// ItemFusing_68436008895082
// MI455X (gfx1250) — matrix operand order repaired and checked by reading, not compiled, not run on hardware
//
#include <hip/hip_runtime.h>

#define T_TOK 131072
#define HI_DIM 1024
#define H_DIM 128
#define NSESS 4096

typedef __attribute__((ext_vector_type(16))) __bf16 bf16x16;
typedef __attribute__((ext_vector_type(8)))  float  f32x8;

union FragB16 { uint4 u[2]; bf16x16 v; };

__device__ __forceinline__ f32x8 wmma_bf16(bf16x16 a, bf16x16 b, f32x8 c) {
  return __builtin_amdgcn_wmma_f32_16x16x32_bf16(false, a, false, b, (short)0, c, false, false);
}
__device__ __forceinline__ float sigmoidf_(float x) { return 1.0f / (1.0f + __expf(-x)); }

// CDNA5 async global->LDS copy (16B per lane), tracked by ASYNCcnt.
__device__ __forceinline__ void async_copy_b128(unsigned lds_off, const void* gaddr) {
  asm volatile("global_load_async_to_lds_b128 %0, %1, off"
               :: "v"(lds_off), "v"(gaddr) : "memory");
}
__device__ __forceinline__ void wait_async0() {
  asm volatile("s_wait_asynccnt 0" ::: "memory");
}
__device__ __forceinline__ unsigned lds_addr_of(const void* p) {
  return (unsigned)(size_t)p;   // LDS aperture: low 32 bits = LDS byte address
}

// build a 16x32 bf16 A-fragment from an fp32 LDS row (K-chunk base already applied)
__device__ __forceinline__ bf16x16 load_cvt_fragA(const float* row, int g) {
  float4 x0 = *(const float4*)(row + 16 * g);
  float4 x1 = *(const float4*)(row + 16 * g + 4);
  float4 x2 = *(const float4*)(row + 16 * g + 8);
  float4 x3 = *(const float4*)(row + 16 * g + 12);
  bf16x16 a;
  a[0]  = (__bf16)x0.x; a[1]  = (__bf16)x0.y; a[2]  = (__bf16)x0.z; a[3]  = (__bf16)x0.w;
  a[4]  = (__bf16)x1.x; a[5]  = (__bf16)x1.y; a[6]  = (__bf16)x1.z; a[7]  = (__bf16)x1.w;
  a[8]  = (__bf16)x2.x; a[9]  = (__bf16)x2.y; a[10] = (__bf16)x2.z; a[11] = (__bf16)x2.w;
  a[12] = (__bf16)x3.x; a[13] = (__bf16)x3.y; a[14] = (__bf16)x3.z; a[15] = (__bf16)x3.w;
  return a;
}

// ---------------------------------------------------------------------------
// Kernel 1: convert Wf1 [128,1024], Wf2 [128,1024], W2 [128,128] fp32 -> bf16
// ---------------------------------------------------------------------------
__global__ __launch_bounds__(256) void convert_weights_kernel(
    const float* __restrict__ wf1, const float* __restrict__ wf2,
    const float* __restrict__ w2,
    __bf16* __restrict__ o1, __bf16* __restrict__ o2, __bf16* __restrict__ o3)
{
  int i = blockIdx.x * 256 + threadIdx.x;
  if (i < H_DIM * HI_DIM) {
    o1[i] = (__bf16)wf1[i];
    o2[i] = (__bf16)wf2[i];
  }
  if (i < H_DIM * H_DIM) o3[i] = (__bf16)w2[i];
}

// ---------------------------------------------------------------------------
// Kernel 2: fully fused. One block = 128 token rows = 4 sessions.
// Double-buffered async-DMA pipeline, one barrier per K-step.
// ---------------------------------------------------------------------------
#define BUF_BYTES 57344   // per double-buffer: X1f(18432) X2f(18432) Wa(10240) Wb(10240)
__global__ __launch_bounds__(256) void fused_session_kernel(
    const float* __restrict__ X1, const float* __restrict__ X2,
    const __bf16* __restrict__ Wf1bf, const __bf16* __restrict__ Wf2bf,
    const __bf16* __restrict__ W2bf,
    const float* __restrict__ Wf1_b, const float* __restrict__ Wf2_b,
    const float* __restrict__ W1_w,  const float* __restrict__ W1_b,
    const float* __restrict__ W2_b,
    const float* __restrict__ q_w,   const float* __restrict__ q_b,
    const float* __restrict__ W3_w,  const float* __restrict__ W3_b,
    float* __restrict__ out)
{
  extern __shared__ char smem[];
  // phase 2 layout (aliased over the phase-1 double buffers after a barrier)
  float*  sH   = (float*)smem;                    // [128][132] fp32 h
  __bf16* sW2r = (__bf16*)(smem + 67584);         // [128][136] resident W2 bf16
  float*  sVn  = (float*)(smem + 102400);         // [4][128]
  float*  sW1v = (float*)(smem + 104448);         // [4][128]
  float*  sAl  = (float*)(smem + 106496);         // [4][32]
  float*  sSg  = (float*)(smem + 107008);         // [4][128]

  const int tid  = threadIdx.x;
  const int wave = tid >> 5;
  const int lane = tid & 31;
  const int g    = lane >> 4;   // half-wave selector
  const int l15  = lane & 15;
  const int row0 = blockIdx.x * 128;

  const f32x8 zero8 = {0.f, 0.f, 0.f, 0.f, 0.f, 0.f, 0.f, 0.f};
  f32x8 acc1[8], acc2[8];
#pragma unroll
  for (int i = 0; i < 8; i++) { acc1[i] = zero8; acc2[i] = zero8; }

  // per-thread DMA coordinates (loop-invariant)
  const int xr0 = tid >> 3, xc0 = tid & 7;   // X: 16B chunk (row, col/4)
  const int wr0 = tid >> 2, wc0 = tid & 3;   // W: 16B chunk (row, col/8)

  // issue the 12 per-thread async copies for K-chunk ks into buffer `buf`
  auto issue_dma = [&](int ks, int buf) {
    char* base = smem + buf * BUF_BYTES;
    unsigned lX1 = lds_addr_of(base);
    unsigned lX2 = lX1 + 18432;
    unsigned lWa = lX1 + 36864;
    unsigned lWb = lX1 + 47104;
#pragma unroll
    for (int i = 0; i < 4; i++) {
      int id = tid + i * 256; int r = id >> 3; int c = id & 7;
      size_t goff = (size_t)(row0 + r) * HI_DIM + ks * 32 + c * 4;
      unsigned loff = (unsigned)(r * 36 + c * 4) * 4u;     // fp32 rows padded to 36
      async_copy_b128(lX1 + loff, X1 + goff);
      async_copy_b128(lX2 + loff, X2 + goff);
    }
#pragma unroll
    for (int i = 0; i < 2; i++) {
      int r = wr0 + i * 64;
      size_t goff = (size_t)r * HI_DIM + ks * 32 + wc0 * 8;
      unsigned loff = (unsigned)(r * 40 + wc0 * 8) * 2u;   // bf16 rows padded to 40
      async_copy_b128(lWa + loff, Wf1bf + goff);
      async_copy_b128(lWb + loff, Wf2bf + goff);
    }
  };

  // ---- main K loop: f1/f2 GEMMs, K = 1024 in chunks of 32 ----
  issue_dma(0, 0);
  for (int ks = 0; ks < 32; ks++) {
    wait_async0();            // own DMA(ks) landed (overlapped with compute(ks-1))
    __syncthreads();          // tiles visible; old-buffer readers retired
    if (ks + 1 < 32) issue_dma(ks + 1, (ks + 1) & 1);   // overlaps compute(ks)

    char* base = smem + (ks & 1) * BUF_BYTES;
    const float*  cX1 = (const float*)base;
    const float*  cX2 = (const float*)(base + 18432);
    const __bf16* cWa = (const __bf16*)(base + 36864);
    const __bf16* cWb = (const __bf16*)(base + 47104);

    bf16x16 a1 = load_cvt_fragA(cX1 + (wave * 16 + l15) * 36, g);
    bf16x16 a2 = load_cvt_fragA(cX2 + (wave * 16 + l15) * 36, g);
#pragma unroll
    for (int nt = 0; nt < 8; nt++) {
      FragB16 b1, b2;
      const __bf16* br1 = cWa + (nt * 16 + l15) * 40 + 16 * g;
      const __bf16* br2 = cWb + (nt * 16 + l15) * 40 + 16 * g;
      b1.u[0] = *(const uint4*)(br1);
      b1.u[1] = *(const uint4*)(br1 + 8);
      b2.u[0] = *(const uint4*)(br2);
      b2.u[1] = *(const uint4*)(br2 + 8);
      acc1[nt] = wmma_bf16(a1, b1.v, acc1[nt]);
      acc2[nt] = wmma_bf16(a2, b2.v, acc2[nt]);
    }
  }

  __syncthreads();            // staging buffers dead; switch to phase-2 layout

  // ---- resident W2 tile: async DMA (raw bf16), overlapped with gate math ----
#pragma unroll
  for (int i = 0; i < 8; i++) {
    int idx = tid + i * 256; int r = idx >> 4; int c = idx & 15;
    async_copy_b128(lds_addr_of(sW2r) + (unsigned)(r * 136 + c * 8) * 2u,
                    W2bf + (size_t)r * H_DIM + c * 8);
  }

  // ---- gate + h (fp32 into LDS) ----
#pragma unroll
  for (int nt = 0; nt < 8; nt++) {
    int n = nt * 16 + l15;
    float bb1 = Wf1_b[n], bb2 = Wf2_b[n];
#pragma unroll
    for (int r = 0; r < 8; r++) {
      float f1v = acc1[nt][r] + bb1;
      float f2v = acc2[nt][r] + bb2;
      float gg  = sigmoidf_(f1v + f2v);
      float h   = f2v + (f1v - f2v) * gg;
      sH[(wave * 16 + 8 * g + r) * 132 + n] = h;
    }
  }
  wait_async0();
  __syncthreads();

  // ---- v_n and w1v = W1 @ v_n + b ----
  for (int it = 0; it < 2; it++) {
    int item = tid + it * 256; int s = item >> 7; int n = item & 127;
    sVn[s * 128 + n] = sH[(s * 32 + 31) * 132 + n];
  }
  __syncthreads();
  for (int it = 0; it < 2; it++) {
    int item = tid + it * 256; int s = item >> 7; int n = item & 127;
    float a = W1_b[n];
    const float* wr = W1_w + (size_t)n * H_DIM;
    const float* v  = sVn + s * 128;
    for (int k = 0; k < H_DIM; k++) a += wr[k] * v[k];
    sW1v[s * 128 + n] = a;
  }
  __syncthreads();

  // ---- second GEMM: w2h = h @ W2^T (K=128), accumulators reused ----
#pragma unroll
  for (int i = 0; i < 8; i++) acc1[i] = zero8;
  const int rowh = wave * 16 + l15;
#pragma unroll
  for (int kq = 0; kq < 4; kq++) {
    int k0 = kq * 32;
    bf16x16 a = load_cvt_fragA(sH + rowh * 132 + k0, g);
#pragma unroll
    for (int nt = 0; nt < 8; nt++) {
      FragB16 b;
      const __bf16* br = sW2r + (nt * 16 + l15) * 136 + k0 + 16 * g;
      b.u[0] = *(const uint4*)(br);
      b.u[1] = *(const uint4*)(br + 8);
      acc1[nt] = wmma_bf16(a, b.v, acc1[nt]);
    }
  }

  // ---- alpha: q . sigmoid(w1v + w2h + b2), reduced across the 16 N-lanes ----
  float qn[8], w2bn[8], w1vn[8];
  const int sess = wave >> 1;   // 4 sessions per block, 2 waves per session
#pragma unroll
  for (int nt = 0; nt < 8; nt++) {
    int n = nt * 16 + l15;
    qn[nt]   = q_w[n];
    w2bn[nt] = W2_b[n];
    w1vn[nt] = sW1v[sess * 128 + n];
  }
  float qbv = q_b[0];
#pragma unroll
  for (int r = 0; r < 8; r++) {
    float sacc = 0.f;
#pragma unroll
    for (int nt = 0; nt < 8; nt++)
      sacc += qn[nt] * sigmoidf_(w1vn[nt] + acc1[nt][r] + w2bn[nt]);
    sacc += __shfl_xor(sacc, 1, 16);
    sacc += __shfl_xor(sacc, 2, 16);
    sacc += __shfl_xor(sacc, 4, 16);
    sacc += __shfl_xor(sacc, 8, 16);
    if (l15 == 0) {
      int trow = (wave & 1) * 16 + 8 * g + r;   // token index within session
      sAl[sess * 32 + trow] = sacc + qbv;
    }
  }
  __syncthreads();

  // ---- s_g = sum_t alpha_t * h_t ----
  for (int it = 0; it < 2; it++) {
    int item = tid + it * 256; int s = item >> 7; int n = item & 127;
    float a = 0.f;
    for (int t = 0; t < 32; t++) a += sAl[s * 32 + t] * sH[(s * 32 + t) * 132 + n];
    sSg[s * 128 + n] = a;
  }
  __syncthreads();

  // ---- h_s = W3 @ [v_n ; s_g] + b ----
  for (int it = 0; it < 2; it++) {
    int item = tid + it * 256; int s = item >> 7; int n = item & 127;
    float o = W3_b[n];
    const float* w3r = W3_w + (size_t)n * (2 * H_DIM);
    const float* v   = sVn + s * 128;
    const float* sg  = sSg + s * 128;
    for (int k = 0; k < H_DIM; k++) o += w3r[k] * v[k];
    for (int k = 0; k < H_DIM; k++) o += w3r[H_DIM + k] * sg[k];
    out[(size_t)(blockIdx.x * 4 + s) * H_DIM + n] = o;
  }
}

// ---------------------------------------------------------------------------
extern "C" void kernel_launch(void* const* d_in, const int* in_sizes, int n_in,
                              void* d_out, int out_size, void* d_ws, size_t ws_size,
                              hipStream_t stream) {
  const float* inter = (const float*)d_in[0];
  const float* intra = (const float*)d_in[1];
  // d_in[2] = seq_len (constant L=32, sessions contiguous) — not needed
  const float* Wf1w = (const float*)d_in[3];
  const float* Wf1b = (const float*)d_in[4];
  const float* Wf2w = (const float*)d_in[5];
  const float* Wf2b = (const float*)d_in[6];
  const float* W1w  = (const float*)d_in[7];
  const float* W1b  = (const float*)d_in[8];
  const float* W2w  = (const float*)d_in[9];
  const float* W2b  = (const float*)d_in[10];
  const float* qw   = (const float*)d_in[11];
  const float* qb   = (const float*)d_in[12];
  const float* W3w  = (const float*)d_in[13];
  const float* W3b  = (const float*)d_in[14];
  float* out = (float*)d_out;

  __bf16* wf1bf = (__bf16*)d_ws;                 // 128*1024 bf16
  __bf16* wf2bf = wf1bf + H_DIM * HI_DIM;        // 128*1024 bf16
  __bf16* w2bf  = wf2bf + H_DIM * HI_DIM;        // 128*128  bf16

  convert_weights_kernel<<<(H_DIM * HI_DIM + 255) / 256, 256, 0, stream>>>(
      Wf1w, Wf2w, W2w, wf1bf, wf2bf, w2bf);

  const size_t smem = 2 * BUF_BYTES;             // 114688: phase-1 double buffers dominate
  fused_session_kernel<<<T_TOK / 128, 256, smem, stream>>>(
      inter, intra, wf1bf, wf2bf, w2bf,
      Wf1b, Wf2b, W1w, W1b, W2b, qw, qb, W3w, W3b, out);
}
